// T_LSTM_AE_30159260352641
// MI455X (gfx1250) — matrix operand order repaired and checked by reading, not compiled, not run on hardware
//
#include <hip/hip_runtime.h>
#include <hip/hip_bf16.h>
#include <math.h>

typedef __bf16 bf16;
typedef __attribute__((ext_vector_type(16))) __bf16 v16bf;
typedef __attribute__((ext_vector_type(8)))  __bf16 v8bf;
typedef __attribute__((ext_vector_type(8)))  float  v8f;

#define S_LEN 256
#define BATCH 128
#define HID   512
#define NGATE 2048
#define NBLK  16
#define HB    32      // hidden units per block

// ---------------------------------------------------------------- helpers

__device__ __forceinline__ float fsigmoid(float x) { return 1.0f / (1.0f + __expf(-x)); }

// A fragment (16x32 bf16), ISA layout: lane l holds row M=l&15;
// K elems 0..7 = kb..kb+7, elems 8..15 = kb+16..kb+23, kb = (l>>4)*8.
__device__ __forceinline__ v16bf load_frag_a(const bf16* tile, int ld, int lane) {
  const bf16* p = tile + (lane & 15) * ld + ((lane >> 4) << 3);
  v8bf lo = *(const v8bf*)p;
  v8bf hi = *(const v8bf*)(p + 16);
  return __builtin_shufflevector(lo, hi, 0,1,2,3,4,5,6,7,8,9,10,11,12,13,14,15);
}

// B fragment (32x16 bf16) from a TRANSPOSED weight matrix Bt[n][k]:
// lane l holds column N=l&15; K elems 0..7 = kb..kb+7, elems 8..15 = kb+16..kb+23, kb = (l>>4)*8.
__device__ __forceinline__ v16bf load_frag_bt(const bf16* bt, int ldk, int lane) {
  const bf16* p = bt + (lane & 15) * ldk + ((lane >> 4) << 3);
  v8bf lo = *(const v8bf*)p;
  v8bf hi = *(const v8bf*)(p + 16);
  return __builtin_shufflevector(lo, hi, 0,1,2,3,4,5,6,7,8,9,10,11,12,13,14,15);
}

__device__ __forceinline__ v8f wmma_bf16(v16bf a, v16bf b, v8f c) {
  return __builtin_amdgcn_wmma_f32_16x16x32_bf16(false, a, false, b, (short)0, c, false, false);
}

// Monotonic grid barrier across the 16 persistent workgroups.
__device__ __forceinline__ void grid_barrier(unsigned* cnt, unsigned target) {
  __syncthreads();
  __threadfence();
  if (threadIdx.x == 0) {
    __hip_atomic_fetch_add(cnt, 1u, __ATOMIC_RELEASE, __HIP_MEMORY_SCOPE_AGENT);
    while (__hip_atomic_load(cnt, __ATOMIC_ACQUIRE, __HIP_MEMORY_SCOPE_AGENT) < target)
      __builtin_amdgcn_s_sleep(1);
  }
  __syncthreads();
  __threadfence();
}

// ---------------------------------------------------------------- prep kernels

__global__ void __launch_bounds__(256) zero_kernel(unsigned* p, int n) {
  int i = blockIdx.x * blockDim.x + threadIdx.x;
  if (i < n) p[i] = 0u;
}

// dst[g*K + k] = (bf16) src[k*G + g]
__global__ void __launch_bounds__(256) transpose_bf16_kernel(bf16* dst, const float* src, int K, int G) {
  int g = blockIdx.x * 32 + (threadIdx.x & 31);
  int k = blockIdx.y * 8  + (threadIdx.x >> 5);
  if (g < G && k < K) dst[(size_t)g * K + k] = (bf16)src[(size_t)k * G + g];
}

// Wcombt[g][k] = sum_j Wo[k][j] * W2[j][g]   (K=512, J=64, G=2048), stored transposed
__global__ void __launch_bounds__(256) wcomb_kernel(bf16* dst, const float* Wo, const float* W2) {
  int g = blockIdx.x * 64 + (threadIdx.x & 63);
  int k = blockIdx.y * 4  + (threadIdx.x >> 6);
  float s = 0.f;
  #pragma unroll 8
  for (int j = 0; j < 64; j++) s += Wo[k * 64 + j] * W2[(size_t)j * NGATE + g];
  dst[(size_t)g * HID + k] = (bf16)s;
}

// bcomb[g] = b2[g] + sum_j bo[j] * W2[j][g]
__global__ void __launch_bounds__(256) bcomb_kernel(float* dst, const float* bo, const float* W2, const float* b2) {
  int g = blockIdx.x * 256 + threadIdx.x;
  if (g >= NGATE) return;
  float s = b2[g];
  for (int j = 0; j < 64; j++) s += bo[j] * W2[(size_t)j * NGATE + g];
  dst[g] = s;
}

// x [B,S,64]->bf16 [S,B,64] (+reversed); Tmap[s][b] = 1/log(t+e) (+reversed)
__global__ void __launch_bounds__(256) prep_inputs_kernel(const float* x, const float* t,
                                                          bf16* x1, bf16* xrev,
                                                          float* Tm, float* Tmr) {
  int i = blockIdx.x * 256 + threadIdx.x;      // i = s*128 + b
  if (i >= S_LEN * BATCH) return;
  int s = i >> 7, b = i & 127;
  const float* src = x + ((size_t)b * S_LEN + s) * 64;
  bf16* d1 = x1   + (size_t)i * 64;
  bf16* d2 = xrev + ((size_t)(S_LEN - 1 - s) * BATCH + b) * 64;
  #pragma unroll 8
  for (int d = 0; d < 64; d++) { bf16 v = (bf16)src[d]; d1[d] = v; d2[d] = v; }
  float T = 1.0f / __logf(t[(size_t)b * S_LEN + s] + 2.7183f);
  Tm[i] = T;
  Tmr[(S_LEN - 1 - s) * BATCH + b] = T;
}

// ---------------------------------------------------------------- recurrent layer

// Persistent kernel: grid = 16 blocks x 256 threads (8 wave32).
// Block bid owns hidden slice [bid*32, bid*32+32).
// g = xin[s] @ Wx + h @ U + bias ; C_ST = sigmoid(c @ Wd + bd)
__global__ void __launch_bounds__(256, 1)
tlstm_layer_kernel(const bf16* __restrict__ xin, int Kx,
                   const bf16* __restrict__ Ut,   // [2048][512] transposed bf16
                   const bf16* __restrict__ Wxt,  // [2048][Kx]  transposed bf16
                   const bf16* __restrict__ Wdt,  // [512][512]  transposed bf16
                   const float* __restrict__ bias,  // [2048]
                   const float* __restrict__ bd,    // [512]
                   const float* __restrict__ Tmap,  // [S*B]
                   const float* __restrict__ cinit, // lane-layout fp32, or null
                   const bf16*  __restrict__ hinit, // [B*512] bf16, or null
                   bf16* __restrict__ hping,        // [2][B*512]
                   bf16* __restrict__ cping,        // [2][B*512]
                   bf16* __restrict__ hs_out,       // [S*B*512] or null
                   float* __restrict__ csave,       // lane-layout fp32, or null
                   unsigned* __restrict__ barrier_cnt)
{
  extern __shared__ bf16 smem[];
  bf16* Ul  = smem;                       // [128 cols][512]  col = q*32 + j
  bf16* Wdl = smem + 128 * 512;           // [32 cols][512]
  bf16* Wxl = smem + 128 * 512 + 32 * 512;// [128 cols][Kx]

  const int bid  = blockIdx.x;
  const int tid  = threadIdx.x;
  const int lane = tid & 31;
  const int w    = tid >> 5;      // wave 0..7
  const int jt   = w & 1;         // hidden 16-range within slice
  const int mq   = w >> 1;        // M quarter (32 rows)
  const int m0   = mq * 32;
  const int nloc = jt * 16 + (lane & 15);
  const int nglob = bid * HB + nloc;

  // ---- stage weight slices into LDS (once) ----
  for (int idx = tid; idx < 128 * 64; idx += 256) {     // U slice: 128 cols x 512 k
    int col = idx >> 6; int kc = (idx & 63) << 3;
    int q = col >> 5, j = col & 31;
    *(v8bf*)(Ul + (size_t)col * 512 + kc) =
        *(const v8bf*)(Ut + ((size_t)(q * 512 + bid * HB + j)) * 512 + kc);
  }
  for (int idx = tid; idx < 32 * 64; idx += 256) {      // Wd slice: 32 cols x 512 k
    int col = idx >> 6; int kc = (idx & 63) << 3;
    *(v8bf*)(Wdl + (size_t)col * 512 + kc) =
        *(const v8bf*)(Wdt + ((size_t)(bid * HB + col)) * 512 + kc);
  }
  const int kxc = Kx >> 3;
  for (int idx = tid; idx < 128 * kxc; idx += 256) {    // Wx slice: 128 cols x Kx k
    int col = idx / kxc; int kc = (idx % kxc) << 3;
    int q = col >> 5, j = col & 31;
    *(v8bf*)(Wxl + (size_t)col * Kx + kc) =
        *(const v8bf*)(Wxt + ((size_t)(q * 512 + bid * HB + j)) * Kx + kc);
  }

  // ---- persistent fp32 cell state in WMMA C layout ----
  v8f creg[2];
  if (cinit) {
    const float* cp = cinit + (((size_t)(bid * 8 + w) * 32 + lane) * 16);
    #pragma unroll
    for (int mt = 0; mt < 2; mt++)
      #pragma unroll
      for (int e = 0; e < 8; e++) creg[mt][e] = cp[mt * 8 + e];
  } else {
    #pragma unroll
    for (int mt = 0; mt < 2; mt++)
      #pragma unroll
      for (int e = 0; e < 8; e++) creg[mt][e] = 0.f;
  }

  // biases for this lane's column (constant across steps)
  float bv0 = bias[0 * 512 + nglob];
  float bv1 = bias[1 * 512 + nglob];
  float bv2 = bias[2 * 512 + nglob];
  float bv3 = bias[3 * 512 + nglob];
  float bdv = bd[nglob];

  // ---- publish initial h / c to ping buffer 0 ----
  #pragma unroll
  for (int mt = 0; mt < 2; mt++)
    #pragma unroll
    for (int e = 0; e < 8; e++) {
      int m = m0 + mt * 16 + e + ((lane >> 4) << 3);
      bf16 hv = hinit ? hinit[(size_t)m * HID + nglob] : (bf16)0.f;
      hping[(size_t)m * HID + nglob] = hv;
      cping[(size_t)m * HID + nglob] = (bf16)creg[mt][e];
    }
  unsigned bar = 1;
  grid_barrier(barrier_cnt, NBLK * bar); bar++;

  const int KXI = Kx >> 5;

  for (int s = 0; s < S_LEN; s++) {
    const bf16* hprev = hping + (size_t)(s & 1) * (BATCH * HID);
    const bf16* cprev = cping + (size_t)(s & 1) * (BATCH * HID);
    bf16* hnext = hping + (size_t)((s + 1) & 1) * (BATCH * HID);
    bf16* cnext = cping + (size_t)((s + 1) & 1) * (BATCH * HID);
    const bf16* xs = xin + (size_t)s * BATCH * Kx;

    if (s + 1 < S_LEN)  // warm L2 with next step's input tile
      __builtin_prefetch(xs + (size_t)BATCH * Kx + (size_t)tid * 64, 0, 1);

    v8f acc[4][2], accd[2];
    #pragma unroll
    for (int mt = 0; mt < 2; mt++)
      #pragma unroll
      for (int e = 0; e < 8; e++) {
        acc[0][mt][e] = bv0; acc[1][mt][e] = bv1;
        acc[2][mt][e] = bv2; acc[3][mt][e] = bv3;
        accd[mt][e] = bdv;
      }

    // h @ U  and  c @ Wd   (K = 512)
    #pragma unroll 4
    for (int kk = 0; kk < 16; kk++) {
      int k0 = kk * 32;
      v16bf a0 = load_frag_a(hprev + (size_t)m0 * HID + k0, HID, lane);
      v16bf a1 = load_frag_a(hprev + (size_t)(m0 + 16) * HID + k0, HID, lane);
      #pragma unroll
      for (int q = 0; q < 4; q++) {
        v16bf bfr = load_frag_bt(Ul + (size_t)(q * 32 + jt * 16) * 512 + k0, 512, lane);
        acc[q][0] = wmma_bf16(a0, bfr, acc[q][0]);
        acc[q][1] = wmma_bf16(a1, bfr, acc[q][1]);
      }
      v16bf c0 = load_frag_a(cprev + (size_t)m0 * HID + k0, HID, lane);
      v16bf c1 = load_frag_a(cprev + (size_t)(m0 + 16) * HID + k0, HID, lane);
      v16bf bd0 = load_frag_bt(Wdl + (size_t)(jt * 16) * 512 + k0, 512, lane);
      accd[0] = wmma_bf16(c0, bd0, accd[0]);
      accd[1] = wmma_bf16(c1, bd0, accd[1]);
    }
    // xin[s] @ Wx  (K = Kx)
    for (int kk = 0; kk < KXI; kk++) {
      int k0 = kk * 32;
      v16bf a0 = load_frag_a(xs + (size_t)m0 * Kx + k0, Kx, lane);
      v16bf a1 = load_frag_a(xs + (size_t)(m0 + 16) * Kx + k0, Kx, lane);
      #pragma unroll
      for (int q = 0; q < 4; q++) {
        v16bf bfr = load_frag_bt(Wxl + (size_t)(q * 32 + jt * 16) * Kx + k0, Kx, lane);
        acc[q][0] = wmma_bf16(a0, bfr, acc[q][0]);
        acc[q][1] = wmma_bf16(a1, bfr, acc[q][1]);
      }
    }

    // elementwise T-LSTM update
    const float* Ts = Tmap + (size_t)s * BATCH;
    #pragma unroll
    for (int mt = 0; mt < 2; mt++) {
      #pragma unroll
      for (int e = 0; e < 8; e++) {
        int m = m0 + mt * 16 + e + ((lane >> 4) << 3);
        float T   = Ts[m];
        float cst = fsigmoid(accd[mt][e]);
        float cadj = creg[mt][e] - cst + T * cst;
        float ig = fsigmoid(acc[0][mt][e]);
        float fg = fsigmoid(acc[1][mt][e]);
        float og = fsigmoid(acc[2][mt][e]);
        float cd = tanhf(acc[3][mt][e]);
        float cn = fg * cadj + ig * cd;
        creg[mt][e] = cn;
        float hv = og * tanhf(cn);
        bf16 hb = (bf16)hv;
        hnext[(size_t)m * HID + nglob] = hb;
        cnext[(size_t)m * HID + nglob] = (bf16)cn;
        if (hs_out) hs_out[((size_t)s * BATCH + m) * HID + nglob] = hb;
      }
    }
    grid_barrier(barrier_cnt, NBLK * bar); bar++;
  }

  if (csave) {
    float* cp = csave + (((size_t)(bid * 8 + w) * 32 + lane) * 16);
    #pragma unroll
    for (int mt = 0; mt < 2; mt++)
      #pragma unroll
      for (int e = 0; e < 8; e++) cp[mt * 8 + e] = creg[mt][e];
  }
}

// ---------------------------------------------------------------- output GEMM
// out[b][255-srev][o] = hsd2[srev][b][:] @ Wo3 + bo3 ; one 16x16 tile per wave.
__global__ void __launch_bounds__(256) out_gemm_kernel(float* __restrict__ out,
                                                       const bf16* __restrict__ hsd2,
                                                       const bf16* __restrict__ Wo3t,
                                                       const float* __restrict__ bo3) {
  int lane = threadIdx.x & 31;
  int wid  = blockIdx.x * 8 + (threadIdx.x >> 5);  // 8192 waves total
  int mt = wid >> 2;        // 0..2047 (rows of [S*B])
  int nt = wid & 3;         // 0..3   (16-col tiles of 64)
  int m0 = mt * 16, n0 = nt * 16;

  v8f acc;
  float bv = bo3[n0 + (lane & 15)];
  #pragma unroll
  for (int e = 0; e < 8; e++) acc[e] = bv;

  #pragma unroll 4
  for (int kk = 0; kk < 16; kk++) {
    int k0 = kk * 32;
    v16bf a = load_frag_a(hsd2 + (size_t)m0 * HID + k0, HID, lane);
    v16bf b = load_frag_bt(Wo3t + (size_t)n0 * HID + k0, HID, lane);
    acc = wmma_bf16(a, b, acc);
  }
  #pragma unroll
  for (int e = 0; e < 8; e++) {
    int m = m0 + e + ((lane >> 4) << 3);
    int srev = m >> 7, b = m & 127;
    int s_out = S_LEN - 1 - srev;
    out[(((size_t)b * S_LEN + s_out) * 64) + n0 + (lane & 15)] = acc[e];
  }
}

// ---------------------------------------------------------------- launch

extern "C" void kernel_launch(void* const* d_in, const int* in_sizes, int n_in,
                              void* d_out, int out_size, void* d_ws, size_t ws_size,
                              hipStream_t stream) {
  const float* input_seq = (const float*)d_in[0];
  const float* time_seq  = (const float*)d_in[1];
  const float* W_enc  = (const float*)d_in[2];
  const float* U_enc  = (const float*)d_in[3];
  const float* b_enc  = (const float*)d_in[4];
  const float* Wd_enc = (const float*)d_in[5];
  const float* bd_enc = (const float*)d_in[6];
  const float* W_enc2  = (const float*)d_in[7];
  const float* U_enc2  = (const float*)d_in[8];
  const float* b_enc2  = (const float*)d_in[9];
  const float* Wd_enc2 = (const float*)d_in[10];
  const float* bd_enc2 = (const float*)d_in[11];
  const float* W_dec  = (const float*)d_in[12];
  const float* U_dec  = (const float*)d_in[13];
  const float* b_dec  = (const float*)d_in[14];
  const float* Wd_dec = (const float*)d_in[15];
  const float* bd_dec = (const float*)d_in[16];
  const float* W_dec2  = (const float*)d_in[17];
  const float* U_dec2  = (const float*)d_in[18];
  const float* b_dec2  = (const float*)d_in[19];
  const float* Wd_dec2 = (const float*)d_in[20];
  const float* bd_dec2 = (const float*)d_in[21];
  const float* Wo  = (const float*)d_in[22];
  const float* bo  = (const float*)d_in[23];
  const float* Wo2 = (const float*)d_in[24];
  const float* bo2 = (const float*)d_in[25];
  const float* Wo3 = (const float*)d_in[26];
  const float* bo3 = (const float*)d_in[27];
  float* out = (float*)d_out;

  char* ws = (char*)d_ws;
  size_t off = 0;
  auto alloc = [&](size_t bytes) -> void* {
    void* r = ws + off; off = (off + bytes + 255) & ~(size_t)255; return r;
  };

  const size_t SZ_UT  = (size_t)NGATE * HID * sizeof(bf16);
  const size_t SZ_WDT = (size_t)HID * HID * sizeof(bf16);
  const size_t SZ_WXT = (size_t)NGATE * 64 * sizeof(bf16);
  const size_t SZ_X   = (size_t)S_LEN * BATCH * 64 * sizeof(bf16);
  const size_t SZ_HS  = (size_t)S_LEN * BATCH * HID * sizeof(bf16);
  const size_t SZ_PING= (size_t)2 * BATCH * HID * sizeof(bf16);

  bf16* Ut1 = (bf16*)alloc(SZ_UT);   bf16* Ut2 = (bf16*)alloc(SZ_UT);
  bf16* Ut3 = (bf16*)alloc(SZ_UT);   bf16* Ut4 = (bf16*)alloc(SZ_UT);
  bf16* Wdt1 = (bf16*)alloc(SZ_WDT); bf16* Wdt2 = (bf16*)alloc(SZ_WDT);
  bf16* Wdt3 = (bf16*)alloc(SZ_WDT); bf16* Wdt4 = (bf16*)alloc(SZ_WDT);
  bf16* Wxt1 = (bf16*)alloc(SZ_WXT); bf16* Wxt3 = (bf16*)alloc(SZ_WXT);
  bf16* Wc1  = (bf16*)alloc(SZ_UT);  bf16* Wc2  = (bf16*)alloc(SZ_UT);
  bf16* Wo3t = (bf16*)alloc((size_t)64 * HID * sizeof(bf16));
  float* bc1 = (float*)alloc(NGATE * sizeof(float));
  float* bc2 = (float*)alloc(NGATE * sizeof(float));
  bf16* x1   = (bf16*)alloc(SZ_X);   bf16* xrev = (bf16*)alloc(SZ_X);
  float* Tm  = (float*)alloc((size_t)S_LEN * BATCH * sizeof(float));
  float* Tmr = (float*)alloc((size_t)S_LEN * BATCH * sizeof(float));
  bf16* hs1  = (bf16*)alloc(SZ_HS);
  bf16* hsd  = (bf16*)alloc(SZ_HS);
  bf16* hsd2 = (bf16*)alloc(SZ_HS);
  bf16* hp1 = (bf16*)alloc(SZ_PING); bf16* cp1 = (bf16*)alloc(SZ_PING);
  bf16* hp2 = (bf16*)alloc(SZ_PING); bf16* cp2 = (bf16*)alloc(SZ_PING);
  bf16* hp3 = (bf16*)alloc(SZ_PING); bf16* cp3 = (bf16*)alloc(SZ_PING);
  bf16* hp4 = (bf16*)alloc(SZ_PING); bf16* cp4 = (bf16*)alloc(SZ_PING);
  float* csave = (float*)alloc((size_t)NBLK * 8 * 32 * 16 * sizeof(float));
  unsigned* cnts = (unsigned*)alloc(4 * sizeof(unsigned));

  (void)hipFuncSetAttribute((const void*)tlstm_layer_kernel,
                            hipFuncAttributeMaxDynamicSharedMemorySize, 294912);

  zero_kernel<<<1, 32, 0, stream>>>(cnts, 4);
  prep_inputs_kernel<<<(S_LEN * BATCH) / 256, 256, 0, stream>>>(
      input_seq, time_seq, x1, xrev, Tm, Tmr);

  // weight transposes / conversions to bf16
  transpose_bf16_kernel<<<dim3(64, 64), 256, 0, stream>>>(Ut1, U_enc, 512, 2048);
  transpose_bf16_kernel<<<dim3(64, 64), 256, 0, stream>>>(Ut2, U_enc2, 512, 2048);
  transpose_bf16_kernel<<<dim3(64, 64), 256, 0, stream>>>(Ut3, U_dec, 512, 2048);
  transpose_bf16_kernel<<<dim3(64, 64), 256, 0, stream>>>(Ut4, U_dec2, 512, 2048);
  transpose_bf16_kernel<<<dim3(16, 64), 256, 0, stream>>>(Wdt1, Wd_enc, 512, 512);
  transpose_bf16_kernel<<<dim3(16, 64), 256, 0, stream>>>(Wdt2, Wd_enc2, 512, 512);
  transpose_bf16_kernel<<<dim3(16, 64), 256, 0, stream>>>(Wdt3, Wd_dec, 512, 512);
  transpose_bf16_kernel<<<dim3(16, 64), 256, 0, stream>>>(Wdt4, Wd_dec2, 512, 512);
  transpose_bf16_kernel<<<dim3(64, 8),  256, 0, stream>>>(Wxt1, W_enc, 64, 2048);
  transpose_bf16_kernel<<<dim3(64, 8),  256, 0, stream>>>(Wxt3, W_dec, 64, 2048);
  transpose_bf16_kernel<<<dim3(2, 64),  256, 0, stream>>>(Wo3t, Wo3, 512, 64);
  // folded inter-layer projections
  wcomb_kernel<<<dim3(32, 128), 256, 0, stream>>>(Wc1, Wo,  W_enc2);
  wcomb_kernel<<<dim3(32, 128), 256, 0, stream>>>(Wc2, Wo2, W_dec2);
  bcomb_kernel<<<8, 256, 0, stream>>>(bc1, bo,  W_enc2, b_enc2);
  bcomb_kernel<<<8, 256, 0, stream>>>(bc2, bo2, W_dec2, b_dec2);

  const size_t lds_small = (size_t)(128 * 512 + 32 * 512 + 128 * 64)  * sizeof(bf16);
  const size_t lds_big   = (size_t)(128 * 512 + 32 * 512 + 128 * 512) * sizeof(bf16);

  // Layer 1: encoder TLSTM on raw input (Kx=64)
  tlstm_layer_kernel<<<NBLK, 256, lds_small, stream>>>(
      x1, 64, Ut1, Wxt1, Wdt1, b_enc, bd_enc, Tm,
      nullptr, nullptr, hp1, cp1, hs1, nullptr, cnts + 0);
  // Layer 2: encoder TLSTM on hs1@Wcomb (Kx=512); save final (h,c)
  tlstm_layer_kernel<<<NBLK, 256, lds_big, stream>>>(
      hs1, 512, Ut2, Wc1, Wdt2, bc1, bd_enc2, Tm,
      nullptr, nullptr, hp2, cp2, nullptr, csave, cnts + 1);
  // Layer 3: decoder TLSTM on reversed input, init from (h2,c2) (Kx=64)
  // S=256 even -> final h of layer 2 lives in ping buffer 0 (== hp2 base)
  tlstm_layer_kernel<<<NBLK, 256, lds_small, stream>>>(
      xrev, 64, Ut3, Wxt3, Wdt3, b_dec, bd_dec, Tmr,
      csave, hp2, hp3, cp3, hsd, nullptr, cnts + 2);
  // Layer 4: decoder TLSTM on hsd@Wcomb2 (Kx=512)
  tlstm_layer_kernel<<<NBLK, 256, lds_big, stream>>>(
      hsd, 512, Ut4, Wc2, Wdt4, bc2, bd_dec2, Tmr,
      nullptr, nullptr, hp4, cp4, hsd2, nullptr, cnts + 3);

  // Final projection + time-reversal + [B,S,O] transpose
  out_gemm_kernel<<<1024, 256, 0, stream>>>(out, hsd2, Wo3t, bo3);

  (void)in_sizes; (void)n_in; (void)out_size; (void)ws_size;
}
